// MultiHeadAttention_26688926777452
// MI455X (gfx1250) — matrix operand order repaired and checked by reading, not compiled, not run on hardware
//
#include <hip/hip_runtime.h>
#include <math.h>

#define BDIM   2
#define HEADS  8
#define NSEQ   4096
#define DHEAD  64
#define EMB    512
// 1/sqrt(64) * log2(e): softmax done in exp2 domain
#define QSCALE (0.125f * 1.44269504088896340736f)
#define BQ     64                // query rows per block (16 per wave)
#define BK     128               // keys per inner iteration
#define NSUB   (BK / 16)         // 8 key-subtiles
#define NCH    (BK / 32)         // 4 k-chunks for PV
#define NWAVES 4
#define KITER  (NSEQ / BK)       // 32
#define KVELEMS (BDIM * HEADS * NSEQ * DHEAD)   // 4,194,304 per tensor

typedef __attribute__((ext_vector_type(16))) __bf16         v16bf;
typedef __attribute__((ext_vector_type(8)))  float          v8f;
typedef __attribute__((ext_vector_type(16))) unsigned short v16us;
typedef __attribute__((ext_vector_type(8)))  unsigned short v8us;

__device__ __forceinline__ unsigned short f2bf(float f) {
  unsigned u = __builtin_bit_cast(unsigned, f);
  u += 0x7FFFu + ((u >> 16) & 1u);          // round-to-nearest-even
  return (unsigned short)(u >> 16);
}
__device__ __forceinline__ unsigned pk2bf(float a, float b) {
#if __has_builtin(__builtin_amdgcn_cvt_pk_bf16_f32)
  typedef __attribute__((ext_vector_type(2))) __bf16 v2bf;
  v2bf p = __builtin_amdgcn_cvt_pk_bf16_f32(a, b);
  return __builtin_bit_cast(unsigned, p);
#else
  return (unsigned)f2bf(a) | ((unsigned)f2bf(b) << 16);
#endif
}

// DPP lane shuffle, confined to 16-lane rows (exactly our C-layout row groups)
template <int CTRL>
__device__ __forceinline__ float dppf(float v) {
  return __builtin_bit_cast(
      float, __builtin_amdgcn_update_dpp(0, __builtin_bit_cast(int, v), CTRL,
                                         0xF, 0xF, true));
}
__device__ __forceinline__ float redmax16(float v) {
  v = fmaxf(v, dppf<0xB1>(v));    // quad_perm [1,0,3,2]  : xor 1
  v = fmaxf(v, dppf<0x4E>(v));    // quad_perm [2,3,0,1]  : xor 2
  v = fmaxf(v, dppf<0x141>(v));   // ROW_HALF_MIRROR
  v = fmaxf(v, dppf<0x140>(v));   // ROW_MIRROR
  return v;
}

// CDNA5 async copy: global -> LDS, 16B per lane, tracked by ASYNCcnt
__device__ __forceinline__ void async_ld_b128(unsigned lds_byte_off, const void* g) {
  asm volatile("global_load_async_to_lds_b128 %0, %1, off"
               :: "v"(lds_byte_off), "v"(g) : "memory");
}

// ---- pre-pass: K -> bf16 [bh][n][d]; V -> bf16 transposed [bh][d][n'] with
// keys permuted within each 128-tile: pos = (j%16)*8 + j/16 (matches P staging) ----
__global__ __launch_bounds__(256)
void cvt_kv(const float* __restrict__ K, const float* __restrict__ V,
            unsigned short* __restrict__ Kbf, unsigned short* __restrict__ VTbf) {
  unsigned idx = blockIdx.x * 256 + threadIdx.x;   // < KVELEMS
  int d  = idx & (DHEAD - 1);
  int n  = (idx >> 6) & (NSEQ - 1);
  int bh = idx >> 18;
  int b  = bh >> 3, h = bh & 7;
  size_t src = (size_t)(b * NSEQ + n) * EMB + h * DHEAD + d;
  Kbf[idx] = f2bf(K[src]);
  int j   = n & (BK - 1);
  int pos = ((j & 15) << 3) | (j >> 4);
  VTbf[((size_t)bh * DHEAD + d) * NSEQ + (n & ~(BK - 1)) + pos] = f2bf(V[src]);
}

__global__ __launch_bounds__(128, 1)
void fa_fwd_wmma(const float* __restrict__ Q,
                 const unsigned short* __restrict__ Kbf,
                 const unsigned short* __restrict__ VTbf,
                 float* __restrict__ O) {
  __shared__ __attribute__((aligned(32))) unsigned short Klds[2][BK * DHEAD];   // 2x16KB
  __shared__ __attribute__((aligned(32))) unsigned short VTlds[2][DHEAD * BK];  // 2x16KB
  __shared__ __attribute__((aligned(32))) unsigned short Pst[NWAVES * 16 * BK]; // 16KB

  const int tid  = threadIdx.x;
  const int wv   = tid >> 5;
  const int lane = tid & 31;
  const int half = lane >> 4;
  const int ln16 = lane & 15;

  const int bid = blockIdx.x;
  const int qb  = bid & 63;            // NSEQ/BQ = 64
  const int h   = (bid >> 6) & 7;
  const int b   = bid >> 9;
  const int bh  = b * HEADS + h;
  const int q0  = qb * BQ + wv * 16;

  const unsigned short* Kbase = Kbf  + (size_t)bh * NSEQ * DHEAD;
  const unsigned short* Vbase = VTbf + (size_t)bh * DHEAD * NSEQ;

  // ---- Q A-fragments (bf16, scale*log2e folded in) ----
  v16bf aq[2];
  {
    const float* qrow = Q + (size_t)(b * NSEQ + q0 + ln16) * EMB + h * DHEAD;
#pragma unroll
    for (int c = 0; c < 2; ++c) {
      v16us u;
#pragma unroll
      for (int i = 0; i < 16; ++i) {
        int k = c * 32 + half * 16 + i;
        u[i] = f2bf(qrow[k] * QSCALE);
      }
      aq[c] = __builtin_bit_cast(v16bf, u);
    }
  }
  // all-ones bf16 B-fragment for WMMA row-sum reduction
  v16bf ones;
  {
    v16us u;
#pragma unroll
    for (int i = 0; i < 16; ++i) u[i] = 0x3F80;
    ones = __builtin_bit_cast(v16bf, u);
  }

  v8f acc[4];
#pragma unroll
  for (int t = 0; t < 4; ++t)
#pragma unroll
    for (int r = 0; r < 8; ++r) acc[t][r] = 0.0f;
  float mrow[8], lrow[8];
#pragma unroll
  for (int r = 0; r < 8; ++r) { mrow[r] = -INFINITY; lrow[r] = 0.0f; }

  // async 16B-chunk tile fill: 1024 chunks per 16KB tile, 8 per thread per tensor
  auto issue_tile = [&](int kb, int buf) {
#pragma unroll
    for (int i = 0; i < 8; ++i) {
      int c = tid + i * 128;
      // K tile: 128 rows x 64 dims (8 chunks per row)
      async_ld_b128((unsigned)(uintptr_t)&Klds[buf][(c >> 3) * DHEAD + (c & 7) * 8],
                    Kbase + (size_t)(kb * BK + (c >> 3)) * DHEAD + (c & 7) * 8);
      // VT tile: 64 dims x 128 positions (16 chunks per row)
      async_ld_b128((unsigned)(uintptr_t)&VTlds[buf][(c >> 4) * BK + (c & 15) * 8],
                    Vbase + (size_t)(c >> 4) * NSEQ + kb * BK + (c & 15) * 8);
    }
  };

  issue_tile(0, 0);

  for (int kb = 0; kb < KITER; ++kb) {
    const int cur = kb & 1;
    asm volatile("s_wait_asynccnt 0" ::: "memory");
    __syncthreads();
    if (kb + 1 < KITER) issue_tile(kb + 1, cur ^ 1);

    // ---- S = Q K^T : eight 16-key subtiles, K=64 -> 2 wmma each ----
    v8f s[NSUB];
#pragma unroll
    for (int nh = 0; nh < NSUB; ++nh) {
      v16bf bk0 = __builtin_bit_cast(
          v16bf, *(const v16us*)&Klds[cur][(nh * 16 + ln16) * DHEAD + half * 16]);
      v8f z = {0.f, 0.f, 0.f, 0.f, 0.f, 0.f, 0.f, 0.f};
      s[nh] = __builtin_amdgcn_wmma_f32_16x16x32_bf16(false, aq[0], false, bk0,
                                                      (short)0, z, false, false);
      v16bf bk1 = __builtin_bit_cast(
          v16bf, *(const v16us*)&Klds[cur][(nh * 16 + ln16) * DHEAD + 32 + half * 16]);
      s[nh] = __builtin_amdgcn_wmma_f32_16x16x32_bf16(false, aq[1], false, bk1,
                                                      (short)0, s[nh], false, false);
    }

    // ---- online softmax in exp2 domain (row m = r + 8*half across 16 lanes) ----
    // score for key j = nh*16 + ln16 lives in s[nh]; staged at pos = ln16*8 + nh,
    // so each lane writes its 8 probs as one b128 store.
    float alpha[8];
#pragma unroll
    for (int r = 0; r < 8; ++r) {
      float mx = fmaxf(fmaxf(fmaxf(s[0][r], s[1][r]), fmaxf(s[2][r], s[3][r])),
                       fmaxf(fmaxf(s[4][r], s[5][r]), fmaxf(s[6][r], s[7][r])));
      mx = redmax16(mx);
      float mnew = fmaxf(mrow[r], mx);
      float a  = __builtin_amdgcn_exp2f(mrow[r] - mnew);
      float p0 = __builtin_amdgcn_exp2f(s[0][r] - mnew);
      float p1 = __builtin_amdgcn_exp2f(s[1][r] - mnew);
      float p2 = __builtin_amdgcn_exp2f(s[2][r] - mnew);
      float p3 = __builtin_amdgcn_exp2f(s[3][r] - mnew);
      float p4 = __builtin_amdgcn_exp2f(s[4][r] - mnew);
      float p5 = __builtin_amdgcn_exp2f(s[5][r] - mnew);
      float p6 = __builtin_amdgcn_exp2f(s[6][r] - mnew);
      float p7 = __builtin_amdgcn_exp2f(s[7][r] - mnew);
      mrow[r]  = mnew;
      alpha[r] = a;
      int m = r + half * 8;
      uint4 w;
      w.x = pk2bf(p0, p1);
      w.y = pk2bf(p2, p3);
      w.z = pk2bf(p4, p5);
      w.w = pk2bf(p6, p7);
      *(uint4*)&Pst[wv * (16 * BK) + m * BK + ln16 * 8] = w;
    }
    // per-wave staging; DS ops of one wave complete in order
    asm volatile("s_wait_dscnt 0" ::: "memory");

    // ---- reload P as four A-fragments (16 rows x 32 positions each) ----
    v16bf pa[NCH];
#pragma unroll
    for (int c = 0; c < NCH; ++c) {
      v8us lo = *(const v8us*)&Pst[wv * (16 * BK) + ln16 * BK + c * 32 + half * 16];
      v8us hi = *(const v8us*)&Pst[wv * (16 * BK) + ln16 * BK + c * 32 + half * 16 + 8];
      v16us u;
#pragma unroll
      for (int i = 0; i < 8; ++i) { u[i] = lo[i]; u[i + 8] = hi[i]; }
      pa[c] = __builtin_bit_cast(v16bf, u);
    }

    // ---- row sums by WMMA: lsum = P * ones (every lane gets its row's sum) ----
    v8f lsum = {0.f, 0.f, 0.f, 0.f, 0.f, 0.f, 0.f, 0.f};
#pragma unroll
    for (int c = 0; c < NCH; ++c)
      lsum = __builtin_amdgcn_wmma_f32_16x16x32_bf16(false, pa[c], false, ones,
                                                     (short)0, lsum, false, false);
#pragma unroll
    for (int r = 0; r < 8; ++r) lrow[r] = lrow[r] * alpha[r] + lsum[r];

    // ---- O = alpha*O + P V : 4 dim-tiles x 4 position-chunks ----
#pragma unroll
    for (int t = 0; t < 4; ++t) {
      v8f c0 = acc[t];
#pragma unroll
      for (int r = 0; r < 8; ++r) c0[r] *= alpha[r];
#pragma unroll
      for (int c = 0; c < NCH; ++c) {
        v16bf bv = __builtin_bit_cast(
            v16bf,
            *(const v16us*)&VTlds[cur][(t * 16 + ln16) * BK + c * 32 + half * 16]);
        c0 = __builtin_amdgcn_wmma_f32_16x16x32_bf16(false, pa[c], false, bv,
                                                     (short)0, c0, false, false);
      }
      acc[t] = c0;
    }
  }

  // ---- normalize and store ----
  float inv[8];
#pragma unroll
  for (int r = 0; r < 8; ++r) inv[r] = 1.0f / lrow[r];
#pragma unroll
  for (int t = 0; t < 4; ++t)
#pragma unroll
    for (int r = 0; r < 8; ++r) {
      int m = r + half * 8;
      size_t idx = (size_t)(b * NSEQ + q0 + m) * EMB + h * DHEAD + t * 16 + ln16;
      O[idx] = acc[t][r] * inv[r];
    }
}

extern "C" void kernel_launch(void* const* d_in, const int* in_sizes, int n_in,
                              void* d_out, int out_size, void* d_ws, size_t ws_size,
                              hipStream_t stream) {
  const float* q = (const float*)d_in[0];
  const float* k = (const float*)d_in[1];
  const float* v = (const float*)d_in[2];
  float* out = (float*)d_out;
  unsigned short* kbf  = (unsigned short*)d_ws;           // 8 MB
  unsigned short* vtbf = kbf + KVELEMS;                   // 8 MB
  cvt_kv<<<dim3(KVELEMS / 256), 256, 0, stream>>>(k, v, kbf, vtbf);
  fa_fwd_wmma<<<dim3(BDIM * HEADS * (NSEQ / BQ)), 128, 0, stream>>>(q, kbf, vtbf, out);
}
